// GAT_8315056685501
// MI455X (gfx1250) — compile-verified, matrix operand order checked by reading, not run on hardware
//
#include <hip/hip_runtime.h>
#include <hip/hip_bf16.h>
#include <stdint.h>

// ---------- types ----------
typedef __attribute__((ext_vector_type(16))) __bf16        v16bf;
typedef __attribute__((ext_vector_type(8)))  float         v8f;
typedef __attribute__((ext_vector_type(4)))  unsigned int  v4u;
typedef __attribute__((ext_vector_type(4)))  float         v4f;

// ---------- helpers ----------
__device__ __forceinline__ unsigned short f2bf(float f) {
    unsigned u = __float_as_uint(f);
    u += 0x7FFFu + ((u >> 16) & 1u);          // round-to-nearest-even
    return (unsigned short)(u >> 16);
}
__device__ __forceinline__ float bf2f(unsigned short h) {
    return __uint_as_float(((unsigned)h) << 16);
}
// Sign-split float atomic max -> native int atomics (no CAS loop).
__device__ __forceinline__ void atomicMaxF(float* addr, float v) {
    if (v >= 0.f) atomicMax((int*)addr, __float_as_int(v));
    else          atomicMin((unsigned int*)addr, __float_as_uint(v));
}

// ---------- elementwise utility kernels ----------
__global__ void k_fill(float* __restrict__ p, float v, int n) {
    int i = blockIdx.x * blockDim.x + threadIdx.x;
    if (i < n) p[i] = v;
}
__global__ void k_f32_to_bf16(const float* __restrict__ in, unsigned short* __restrict__ out, int n) {
    int i = blockIdx.x * blockDim.x + threadIdx.x;
    if (i < n) out[i] = f2bf(in[i]);
}

// ---------- W pack: f32 [K x 256] -> bf16, WMMA B-fragment-ready layout ----------
// B fragment (16-bit, 32x16 KxN): lane -> N=lane&15, lanes 16-31 take K+8;
// vector elements j: K = (j<8 ? j : j+8) + 8*(lane>=16).  One contiguous 32B per lane.
__global__ void k_pack_w(const float* __restrict__ W, unsigned short* __restrict__ Wp,
                         int K, int Nout) {
    int Kt = K >> 5, Nt = Nout >> 4;
    int t = blockIdx.x * blockDim.x + threadIdx.x;
    if (t >= Nt * Kt * 32) return;
    int lane = t & 31, tile = t >> 5;
    int kt = tile % Kt, nt = tile / Kt;
    int n    = nt * 16 + (lane & 15);
    int koff = (lane >> 4) << 3;
    unsigned short* dst = Wp + ((size_t)tile * 32 + lane) * 16;
#pragma unroll
    for (int j = 0; j < 16; ++j) {
        int k = (kt << 5) + (j < 8 ? j : j + 8) + koff;
        dst[j] = f2bf(W[(size_t)k * Nout + n]);
    }
}

// ---------- GEMM via WMMA: Y[M x 256] = Xb[M x K](bf16) * Wp(bf16 packed), f32 acc ----------
// grid(M/16, 2), block 256 (8 waves); wave w -> N-tile (blockIdx.y*8 + w).
template <int K>
__global__ void k_gemm_wmma(const unsigned short* __restrict__ Xb,
                            const unsigned short* __restrict__ Wp,
                            float* __restrict__ Y, int M) {
    constexpr int Kt = K / 32;
    const int lane = threadIdx.x & 31;
    const int wave = threadIdx.x >> 5;
    const int mt = blockIdx.x;
    const int nt = blockIdx.y * 8 + wave;
    const int row = mt * 16 + (lane & 15);
    const int c0  = (lane >> 4) << 3;                 // lanes 16-31: K offset +8
    const v4u* ap = (const v4u*)(Xb + (size_t)row * K + c0);
    const v4u* bp = (const v4u*)(Wp + ((size_t)nt * Kt * 32 + lane) * 16);
    v8f acc = {};
#pragma unroll
    for (int kt = 0; kt < Kt; ++kt) {
        union { v4u q[2]; v16bf v; } A, B;
        A.q[0] = ap[0];  A.q[1] = ap[2];              // elements {c0..c0+7, c0+16..c0+23}
        B.q[0] = bp[0];  B.q[1] = bp[1];
        ap += 4;                                       // +32 bf16
        bp += 64;                                      // +32*16 bf16 tile
        acc = __builtin_amdgcn_wmma_f32_16x16x32_bf16(
            false, A.v, false, B.v, (short)0, acc, false, false);
    }
    // D layout: VGPR r -> M = r + 8*(lane>=16); N = lane&15
    const int col   = nt * 16 + (lane & 15);
    const int rbase = mt * 16 + ((lane >> 4) << 3);
#pragma unroll
    for (int r = 0; r < 8; ++r)
        Y[(size_t)(rbase + r) * 256 + col] = acc[r];
}

// ---------- attention scalars: al_src/al_dst [N x 8] ----------
__global__ void k_attn(const float* __restrict__ xl, const float* __restrict__ asrc,
                       const float* __restrict__ adst, float* __restrict__ als,
                       float* __restrict__ ald, int n) {
    int t = blockIdx.x * blockDim.x + threadIdx.x;
    if (t >= n * 8) return;
    int node = t >> 3, h = t & 7;
    const v4f* xp = (const v4f*)(xl + (size_t)node * 256 + h * 32);
    const v4f* ap = (const v4f*)(asrc + h * 32);
    const v4f* dp = (const v4f*)(adst + h * 32);
    float s0 = 0.f, s1 = 0.f;
#pragma unroll
    for (int c = 0; c < 8; ++c) {
        v4f x = xp[c], a = ap[c], d = dp[c];
        s0 += x.x * a.x + x.y * a.y + x.z * a.z + x.w * a.w;
        s1 += x.x * d.x + x.y * d.y + x.z * d.z + x.w * d.w;
    }
    als[t] = s0; ald[t] = s1;
}

// ---------- edge pass 1: leaky-relu logits + segment max ----------
__global__ void k_edge_max(const long long* __restrict__ ei, int E, int Ntot, int H,
                           const float* __restrict__ als, const float* __restrict__ ald,
                           float* __restrict__ m) {
    int i = blockIdx.x * blockDim.x + threadIdx.x;
    if (i >= Ntot) return;
    int s, d;
    if (i < E) { s = (int)ei[i]; d = (int)ei[(size_t)E + i]; } else { s = d = i - E; }
    for (int h = 0; h < H; ++h) {
        float e = als[(size_t)s * H + h] + ald[(size_t)d * H + h];
        e = e > 0.f ? e : 0.2f * e;
        atomicMaxF(&m[(size_t)d * H + h], e);
    }
}

// ---------- edge pass 2: exp + segment sum (caches per-edge exp) ----------
__global__ void k_edge_expsum(const long long* __restrict__ ei, int E, int Ntot, int H,
                              const float* __restrict__ als, const float* __restrict__ ald,
                              const float* __restrict__ m, float* __restrict__ ex,
                              float* __restrict__ ssum) {
    int i = blockIdx.x * blockDim.x + threadIdx.x;
    if (i >= Ntot) return;
    int s, d;
    if (i < E) { s = (int)ei[i]; d = (int)ei[(size_t)E + i]; } else { s = d = i - E; }
    for (int h = 0; h < H; ++h) {
        float e = als[(size_t)s * H + h] + ald[(size_t)d * H + h];
        e = e > 0.f ? e : 0.2f * e;
        float v = __expf(e - m[(size_t)d * H + h]);
        ex[(size_t)i * H + h] = v;
        atomicAdd(&ssum[(size_t)d * H + h], v);
    }
}

// ---------- edge pass 3: acc[dst] += alpha * xl[src]  (H=8, D=256) ----------
// 64 threads per edge, 4 floats each (vectorized gather + f32 atomics into L2).
__global__ void k_edge_agg(const long long* __restrict__ ei, int E, int Ntot,
                           const float* __restrict__ ex, const float* __restrict__ ssum,
                           const float* __restrict__ xl, float* __restrict__ acc) {
    int t = threadIdx.x;
    int i = blockIdx.x * 4 + (t >> 6);
    if (i >= Ntot) return;
    int s, d;
    if (i < E) { s = (int)ei[i]; d = (int)ei[(size_t)E + i]; } else { s = d = i - E; }
    int off = (t & 63) << 2;
    int h   = off >> 5;
    float alpha = ex[(size_t)i * 8 + h] / (ssum[(size_t)d * 8 + h] + 1e-16f);
    v4f xv = *(const v4f*)(xl + (size_t)s * 256 + off);
    float* op = acc + (size_t)d * 256 + off;
    atomicAdd(op + 0, alpha * xv.x);
    atomicAdd(op + 1, alpha * xv.y);
    atomicAdd(op + 2, alpha * xv.z);
    atomicAdd(op + 3, alpha * xv.w);
}

// ---------- bias + ELU + cast to bf16 for next layer's WMMA ----------
__global__ void k_finalize(const float* __restrict__ acc, const float* __restrict__ bias,
                           unsigned short* __restrict__ hb, int total) {
    int i = blockIdx.x * blockDim.x + threadIdx.x;
    if (i >= total) return;
    float v = acc[i] + bias[i & 255];
    v = v > 0.f ? v : __expf(v) - 1.f;
    hb[i] = f2bf(v);
}

// ---------- layer 3: 256 -> 1 dot per node (one wave32 per node) ----------
__global__ void k_dot256(const unsigned short* __restrict__ hb, const float* __restrict__ w,
                         float* __restrict__ y, int n) {
    int wave = threadIdx.x >> 5, lane = threadIdx.x & 31;
    int node = blockIdx.x * 8 + wave;
    if (node >= n) return;
    const unsigned short* xp = hb + (size_t)node * 256;
    float s = 0.f;
#pragma unroll
    for (int c = lane; c < 256; c += 32) s += bf2f(xp[c]) * w[c];
    for (int o = 16; o > 0; o >>= 1) s += __shfl_down(s, o, 32);
    if (lane == 0) y[node] = s;
}
__global__ void k_scale3(const float* __restrict__ y, const float* __restrict__ a_s,
                         const float* __restrict__ a_d, float* __restrict__ als,
                         float* __restrict__ ald, int n) {
    int i = blockIdx.x * blockDim.x + threadIdx.x;
    if (i >= n) return;
    als[i] = y[i] * a_s[0];
    ald[i] = y[i] * a_d[0];
}
__global__ void k_edge_agg1(const long long* __restrict__ ei, int E, int Ntot,
                            const float* __restrict__ ex, const float* __restrict__ ssum,
                            const float* __restrict__ y, float* __restrict__ out) {
    int i = blockIdx.x * blockDim.x + threadIdx.x;
    if (i >= Ntot) return;
    int s, d;
    if (i < E) { s = (int)ei[i]; d = (int)ei[(size_t)E + i]; } else { s = d = i - E; }
    float alpha = ex[i] / (ssum[d] + 1e-16f);
    atomicAdd(&out[d], alpha * y[s]);
}
__global__ void k_bias3(float* __restrict__ out, const float* __restrict__ b, int n) {
    int i = blockIdx.x * blockDim.x + threadIdx.x;
    if (i < n) out[i] += b[0];
}

// ---------- orchestration ----------
extern "C" void kernel_launch(void* const* d_in, const int* in_sizes, int n_in,
                              void* d_out, int out_size, void* d_ws, size_t ws_size,
                              hipStream_t stream) {
    (void)in_sizes; (void)n_in; (void)out_size; (void)ws_size;
    const int N = 30000, E = 480000, Ntot = E + N;
    const float*     x  = (const float*)d_in[0];
    const long long* ei = (const long long*)d_in[1];
    const float* W[4]  = {(const float*)d_in[2], (const float*)d_in[6],  (const float*)d_in[10], (const float*)d_in[14]};
    const float* As[4] = {(const float*)d_in[3], (const float*)d_in[7],  (const float*)d_in[11], (const float*)d_in[15]};
    const float* Ad[4] = {(const float*)d_in[4], (const float*)d_in[8],  (const float*)d_in[12], (const float*)d_in[16]};
    const float* Bi[4] = {(const float*)d_in[5], (const float*)d_in[9],  (const float*)d_in[13], (const float*)d_in[17]};
    float* out = (float*)d_out;

    char* p = (char*)d_ws;
    auto alloc = [&](size_t bytes) -> char* {
        char* r = p; p += (bytes + 255) & ~(size_t)255; return r;
    };
    unsigned short* hb  = (unsigned short*)alloc((size_t)N * 256 * 2);
    float* xl   = (float*)alloc((size_t)N * 256 * 4);
    float* acc  = (float*)alloc((size_t)N * 256 * 4);
    float* als  = (float*)alloc((size_t)N * 8 * 4);
    float* ald  = (float*)alloc((size_t)N * 8 * 4);
    float* mbuf = (float*)alloc((size_t)N * 8 * 4);
    float* sbuf = (float*)alloc((size_t)N * 8 * 4);
    float* ex   = (float*)alloc((size_t)Ntot * 8 * 4);
    unsigned short* Wp = (unsigned short*)alloc((size_t)256 * 256 * 2);
    float* y3   = (float*)alloc((size_t)N * 4);

    const float NEG_INF = -__builtin_inff();
    dim3 b256(256);
    auto cdiv = [](int a, int b) { return (a + b - 1) / b; };

    for (int L = 0; L < 3; ++L) {
        const int K = (L == 0) ? 64 : 256;
        if (L == 0)
            k_f32_to_bf16<<<cdiv(N * 64, 256), b256, 0, stream>>>(x, hb, N * 64);
        k_pack_w<<<cdiv(16 * (K / 32) * 32, 256), b256, 0, stream>>>(W[L], Wp, K, 256);
        if (L == 0)
            k_gemm_wmma<64><<<dim3(N / 16, 2), b256, 0, stream>>>(hb, Wp, xl, N);
        else
            k_gemm_wmma<256><<<dim3(N / 16, 2), b256, 0, stream>>>(hb, Wp, xl, N);
        k_attn<<<cdiv(N * 8, 256), b256, 0, stream>>>(xl, As[L], Ad[L], als, ald, N);
        k_fill<<<cdiv(N * 8, 256), b256, 0, stream>>>(mbuf, NEG_INF, N * 8);
        k_fill<<<cdiv(N * 8, 256), b256, 0, stream>>>(sbuf, 0.f, N * 8);
        k_fill<<<cdiv(N * 256, 256), b256, 0, stream>>>(acc, 0.f, N * 256);
        k_edge_max<<<cdiv(Ntot, 256), b256, 0, stream>>>(ei, E, Ntot, 8, als, ald, mbuf);
        k_edge_expsum<<<cdiv(Ntot, 256), b256, 0, stream>>>(ei, E, Ntot, 8, als, ald, mbuf, ex, sbuf);
        k_edge_agg<<<cdiv(Ntot, 4), b256, 0, stream>>>(ei, E, Ntot, ex, sbuf, xl, acc);
        k_finalize<<<cdiv(N * 256, 256), b256, 0, stream>>>(acc, Bi[L], hb, N * 256);
    }

    // ---- layer 3 (D -> 1, single head) ----
    k_dot256<<<cdiv(N, 8), b256, 0, stream>>>(hb, W[3], y3, N);
    k_scale3<<<cdiv(N, 256), b256, 0, stream>>>(y3, As[3], Ad[3], als, ald, N);
    k_fill<<<cdiv(N, 256), b256, 0, stream>>>(mbuf, NEG_INF, N);
    k_fill<<<cdiv(N, 256), b256, 0, stream>>>(sbuf, 0.f, N);
    k_fill<<<cdiv(N, 256), b256, 0, stream>>>(out, 0.f, N);
    k_edge_max<<<cdiv(Ntot, 256), b256, 0, stream>>>(ei, E, Ntot, 1, als, ald, mbuf);
    k_edge_expsum<<<cdiv(Ntot, 256), b256, 0, stream>>>(ei, E, Ntot, 1, als, ald, mbuf, ex, sbuf);
    k_edge_agg1<<<cdiv(Ntot, 256), b256, 0, stream>>>(ei, E, Ntot, ex, sbuf, y3, out);
    k_bias3<<<cdiv(N, 256), b256, 0, stream>>>(out, Bi[3], N);
}
